// StyleGAN3Generator_970662609056
// MI455X (gfx1250) — compile-verified, matrix operand order checked by reading, not run on hardware
//
#include <hip/hip_runtime.h>
#include <math.h>

// ---------------------------------------------------------------------------
// StyleGAN generator forward for MI455X (gfx1250, wave32, WMMA + TDM).
//
// Demodulated conv == (x * s[b,in]) conv w_shared, then out * d[b,out].
// All heavy math maps to v_wmma_f32_16x16x32_bf16. Weights and activations
// are pre-converted to bf16 (repack kernels / fused upsample) so GEMM inner
// loops are pure b128 loads + v_wmma. The conv double-buffers LDS and pulls
// its weight tiles with the Tensor Data Mover (tensor_load_to_lds) where the
// builtin is available, synchronized with s_wait_tensorcnt.
// ---------------------------------------------------------------------------

typedef __bf16 bf16;
typedef __attribute__((ext_vector_type(16))) __bf16 v16bf;
typedef __attribute__((ext_vector_type(8)))  __bf16 v8bf;
typedef __attribute__((ext_vector_type(8)))  float  v8f;
typedef __attribute__((ext_vector_type(4)))  unsigned int u32x4;
typedef __attribute__((ext_vector_type(8)))  int i32x8;
typedef __attribute__((ext_vector_type(4)))  int i32x4;

#ifndef HAVE_TDM
#  if defined(__has_builtin)
#    if __has_builtin(__builtin_amdgcn_tensor_load_to_lds) && \
        __has_builtin(__builtin_amdgcn_s_wait_tensorcnt)
#      define HAVE_TDM 1
#    else
#      define HAVE_TDM 0
#    endif
#  else
#    define HAVE_TDM 0
#  endif
#endif

__device__ __forceinline__ float leaky02(float v) { return v >= 0.f ? v : 0.2f * v; }
__device__ __forceinline__ int   iclamp(int v, int lo, int hi) {
  return v < lo ? lo : (v > hi ? hi : v);
}

// Generic f32 -> bf16 repack (weights become WMMA-ready once per launch).
__global__ void f32_to_bf16_kernel(const float* __restrict__ src,
                                   bf16* __restrict__ dst, int n)
{
  const int i = blockIdx.x * blockDim.x + threadIdx.x;
  if (i < n) dst[i] = (bf16)src[i];
}

// ---------------------------------------------------------------------------
// Mapping network: PixelNorm + 8x (EqualLinear + leaky). One workgroup,
// 16 waves, A padded 4 -> 16 rows. EqualLinear scale folded into A so the
// inner loop is: 2x global b128 (bf16 W) + 2x ds b128 (bf16 x) + v_wmma.
// ---------------------------------------------------------------------------
__global__ __launch_bounds__(512)
void mapping_wmma(const float* __restrict__ z,      // [4,512]
                  const bf16*  __restrict__ mapwbf, // [8,512,512] bf16
                  const float* __restrict__ map_b,  // [8,512]
                  float* __restrict__ w_out)        // [4,512]
{
  __shared__ __align__(16) bf16 xs[16][520];   // row stride 1040B (16B mult)
  __shared__ float norms[4];
  const int tid  = threadIdx.x;
  const int wv   = tid >> 5;
  const int lane = tid & 31;
  const float scale = 0.01f / sqrtf(512.f);

  if (tid < 4) {
    float ssum = 0.f;
    for (int i = 0; i < 512; ++i) { float v = z[tid * 512 + i]; ssum += v * v; }
    norms[tid] = sqrtf(ssum / 512.f) + 1e-8f;
  }
  __syncthreads();
  for (int r = 0; r < 16; ++r)
    xs[r][tid] = (bf16)((r < 4) ? z[r * 512 + tid] / norms[r] * scale : 0.f);
  __syncthreads();

  const int m  = lane & 15;
  const int kb = (lane < 16) ? 0 : 8;

  for (int l = 0; l < 8; ++l) {
    const bf16*  Wl = mapwbf + (size_t)l * 512 * 512;
    const float* bl = map_b + l * 512;
    v8f acc0 = {}, acc1 = {};
    for (int t = 0; t < 2; ++t) {
      const int nt = wv + t * 16;
      const int n  = nt * 16 + m;
      v8f acc = {};
      for (int kk = 0; kk < 512; kk += 32) {
        union { v16bf v; v8bf h[2]; } fa, fb;
        fa.h[0] = *(const v8bf*)&xs[m][kk + kb];
        fa.h[1] = *(const v8bf*)&xs[m][kk + 16 + kb];
        fb.h[0] = *(const v8bf*)&Wl[(size_t)n * 512 + kk + kb];
        fb.h[1] = *(const v8bf*)&Wl[(size_t)n * 512 + kk + 16 + kb];
        acc = __builtin_amdgcn_wmma_f32_16x16x32_bf16(false, fa.v, false, fb.v,
                                                      (short)0, acc, false, false);
      }
      if (t == 0) acc0 = acc; else acc1 = acc;
    }
    __syncthreads();
    for (int t = 0; t < 2; ++t) {
      v8f acc = t ? acc1 : acc0;
      const int n    = (wv + t * 16) * 16 + (lane & 15);
      const float bs = bl[n] * 0.01f;
#pragma unroll
      for (int r = 0; r < 8; ++r) {
        const int row = r + ((lane < 16) ? 0 : 8);
        const float v = leaky02(acc[r] + bs);
        if (l < 7)          xs[row][n] = (bf16)(v * scale);  // next layer A
        else if (row < 4)   w_out[row * 512 + n] = v;        // final latents
      }
    }
    __syncthreads();
  }
}

// ---------------------------------------------------------------------------
// Style projection: s[b,n] = dot(w[b,:], mw[n,:])/sqrt(512) + mb[n] + 1.
// Scale folded into A; B read directly as bf16.
// ---------------------------------------------------------------------------
__global__ __launch_bounds__(128)
void style_wmma(const float* __restrict__ w,    // [4,512]
                const bf16*  __restrict__ mwbf, // [C,512] bf16
                const float* __restrict__ mb,   // [C]
                float* __restrict__ s, int C)
{
  __shared__ __align__(16) bf16 ws[16][520];
  const int tid  = threadIdx.x;
  const int wv   = tid >> 5;
  const int lane = tid & 31;
  const float scale = rsqrtf(512.f);

  for (int i = tid; i < 16 * 512; i += 128) {
    const int r = i >> 9, k = i & 511;
    ws[r][k] = (bf16)((r < 4) ? w[r * 512 + k] * scale : 0.f);
  }
  __syncthreads();

  const int nt = blockIdx.x * 4 + wv;
  if (nt * 16 < C) {                          // wave-uniform, after barrier
    const int m  = lane & 15;
    const int kb = (lane < 16) ? 0 : 8;
    const int n  = nt * 16 + m;
    v8f acc = {};
    for (int kk = 0; kk < 512; kk += 32) {
      union { v16bf v; v8bf h[2]; } fa, fb;
      fa.h[0] = *(const v8bf*)&ws[m][kk + kb];
      fa.h[1] = *(const v8bf*)&ws[m][kk + 16 + kb];
      fb.h[0] = *(const v8bf*)&mwbf[(size_t)n * 512 + kk + kb];
      fb.h[1] = *(const v8bf*)&mwbf[(size_t)n * 512 + kk + 16 + kb];
      acc = __builtin_amdgcn_wmma_f32_16x16x32_bf16(false, fa.v, false, fb.v,
                                                    (short)0, acc, false, false);
    }
    const int ncol = nt * 16 + (lane & 15);
#pragma unroll
    for (int r = 0; r < 8; ++r) {
      const int row = r + ((lane < 16) ? 0 : 8);
      if (row < 4 && ncol < C)
        s[row * C + ncol] = acc[r] + mb[ncol] + 1.0f;
    }
  }
}

// ---------------------------------------------------------------------------
// Per-(out,in) squared-weight sums + demod factor.
// ---------------------------------------------------------------------------
__global__ void wsq_kernel(const float* __restrict__ cw, float* __restrict__ wsq, int n)
{
  const int i = blockIdx.x * blockDim.x + threadIdx.x;
  if (i >= n) return;
  float a = 0.f;
#pragma unroll
  for (int t = 0; t < 9; ++t) { const float v = cw[(size_t)i * 9 + t]; a += v * v; }
  wsq[i] = a;
}

__global__ void demod_kernel(const float* __restrict__ wsq,
                             const float* __restrict__ s,
                             float* __restrict__ d, int C, int O)
{
  const int i = blockIdx.x * blockDim.x + threadIdx.x;
  if (i >= 4 * O) return;
  const int b = i / O, o = i % O;
  const float* wr = wsq + (size_t)o * C;
  const float* sr = s + b * C;
  float acc = 0.f;
  for (int c = 0; c < C; ++c) { const float sv = sr[c]; acc += wr[c] * sv * sv; }
  d[i] = rsqrtf(acc / (float)(C * 9));
}

// ---------------------------------------------------------------------------
// Weight repack: [O,C,3,3] f32 -> [tap][O][C] bf16 with 1/sqrt(C*9) folded in.
// ---------------------------------------------------------------------------
__global__ void wrepack_kernel(const float* __restrict__ cw,
                               bf16* __restrict__ wrep, int O, int C)
{
  const int idx = blockIdx.x * blockDim.x + threadIdx.x;
  const int oc  = O * C;
  if (idx >= 9 * oc) return;
  const int tap = idx / oc;
  const int rem = idx - tap * oc;               // o*C + c
  const float wscale = rsqrtf((float)(C * 9));
  wrep[idx] = (bf16)(cw[(size_t)rem * 9 + tap] * wscale);
}

// ---------------------------------------------------------------------------
// Fused bilinear 2x upsample + style modulation + bf16 quantize.
// ---------------------------------------------------------------------------
__global__ void upmod_kernel(const float* __restrict__ xin,  // [4,C,H,W] f32
                             const float* __restrict__ s,    // [4,C]
                             bf16* __restrict__ xm,          // [4,C,2H,2W] bf16
                             int C, int H, int W)
{
  const int Ho = 2 * H, Wo = 2 * W;
  const size_t HWo = (size_t)Ho * Wo;
  const size_t idx = (size_t)blockIdx.x * blockDim.x + threadIdx.x;
  const size_t total = (size_t)4 * C * HWo;
  if (idx >= total) return;
  const int p  = (int)(idx % HWo);
  const size_t pl = idx / HWo;
  const int c = (int)(pl % C), b = (int)(pl / C);
  const int oy = p / Wo, ox = p % Wo;
  const float sy = oy * 0.5f - 0.25f, sx = ox * 0.5f - 0.25f;
  int y0 = (int)floorf(sy), x0 = (int)floorf(sx);
  const float wy = sy - (float)y0, wx = sx - (float)x0;
  int y1 = iclamp(y0 + 1, 0, H - 1), x1 = iclamp(x0 + 1, 0, W - 1);
  y0 = iclamp(y0, 0, H - 1); x0 = iclamp(x0, 0, W - 1);
  const float* src = xin + pl * (size_t)H * W;
  const float v00 = src[y0 * W + x0], v01 = src[y0 * W + x1];
  const float v10 = src[y1 * W + x0], v11 = src[y1 * W + x1];
  const float v = (1.f - wy) * ((1.f - wx) * v00 + wx * v01) +
                  wy        * ((1.f - wx) * v10 + wx * v11);
  xm[idx] = (bf16)(v * s[b * C + c]);
}

// Block 0: broadcast learned const to batch + modulate + bf16.
__global__ void bcastmod_kernel(const float* __restrict__ cnst,  // [C,HW]
                                const float* __restrict__ s,     // [4,C]
                                bf16* __restrict__ xm,           // [4,C,HW]
                                int C, int HW)
{
  const int idx = blockIdx.x * blockDim.x + threadIdx.x;
  if (idx >= 4 * C * HW) return;
  const int b = idx / (C * HW);
  const int rem = idx - b * C * HW;
  const int c = rem / HW;
  xm[idx] = (bf16)(cnst[rem] * s[b * C + c]);
}

// ---------------------------------------------------------------------------
// Modulated 3x3 conv as implicit GEMM (9 shifted 1x1 taps), all-bf16, with
// double-buffered LDS. WG = 8 waves: 2(M) x 4(N) -> 64 pixels x 64 outch,
// each wave does two 16x16 WMMAs per K-step sharing one B fragment.
// B tiles (64 rows x 64B, stride C*2B) are DMA'd by the Tensor Data Mover
// with LDS row padding 16B/64B to match the 80B LDS row stride; fallback is
// cooperative staging. Epilogue fuses demod, noise, bias, leaky-ReLU.
// ---------------------------------------------------------------------------
__global__ __launch_bounds__(256)
void modconv3x3_wmma(const bf16* __restrict__ xm,    // [4,C,H,W] modulated bf16
                     const bf16* __restrict__ wrep,  // [9,O,C] bf16, pre-scaled
                     const float* __restrict__ d,    // [4,O]
                     const float* __restrict__ noise,// [4,1,H,W]
                     const float* __restrict__ noise_w_all, int noise_idx,
                     const float* __restrict__ bias, // [O]
                     float* __restrict__ xout,       // [4,O,H,W]
                     int C, int O, int H, int W)
{
  const int b    = blockIdx.z;
  const int HW   = H * W;
  const int p0   = blockIdx.x * 64;
  const int n0   = blockIdx.y * 64;
  const int tid  = threadIdx.x;
  const int lane = tid & 31;
  const int wv   = tid >> 5;
  const int mwv  = wv & 1;        // M half: rows [mwv*32, mwv*32+32)
  const int nwv  = wv >> 1;       // N quarter: cols [nwv*16, nwv*16+16)

  // Row stride 40 bf16 = 80B (multiple of 16B) keeps v8bf loads aligned and
  // matches the TDM pad (4 DWORDs after every 16 DWORDs).
  __shared__ __align__(16) bf16 sA[2][64][40];
  __shared__ __align__(16) bf16 sB[2][64][40];

  const int kc = C >> 5;          // K chunks per tap
  const int nk = 9 * kc;

  auto stageA = [&](int ks, int buf) {
    const int tap = ks / kc, c0 = (ks - tap * kc) * 32;
    const int dy = tap / 3 - 1, dx = tap % 3 - 1;
    for (int i = tid; i < 2048; i += 256) {
      const int pix = i & 63, c = i >> 6;      // pixel fastest -> coalesced
      const int p = p0 + pix;
      bf16 v = (bf16)0.f;
      if (p < HW) {
        const int y = p / W + dy, x = p % W + dx;
        if (y >= 0 && y < H && x >= 0 && x < W)
          v = xm[((size_t)(b * C + c0 + c)) * HW + y * W + x];
      }
      sA[buf][pix][c] = v;
    }
  };

  auto stageB = [&](int ks, int buf) {
    const int tap = ks / kc, c0 = (ks - tap * kc) * 32;
    const bf16* wt = wrep + (size_t)tap * O * C;
#if HAVE_TDM
    if (wv == 0) {
      // Tensor DMA descriptor (D#), ISA 8.3/8.4 layout: 2D tile 32(c) x 64(o)
      // of 2-byte elements, tensor dims [C, O], row stride C, LDS padding of
      // 4 DWORDs every 16 DWORDs to produce the 80B LDS row pitch.
      const unsigned long long ga =
          (unsigned long long)(const void*)(wt + (size_t)n0 * C + c0);
      const unsigned lds = (unsigned)(unsigned long long)(void*)&sB[buf][0][0];
      u32x4 g0 = { 1u,                                  // count=1 (valid D#)
                   lds,                                 // lds_addr
                   (unsigned)ga,                        // global_addr[31:0]
                   (unsigned)(ga >> 32) | (2u << 30) }; // addr[56:32] | type=2
      i32x8 g1 = { (int)((1u << 16)  |                  // data_size = 2B
                         (1u << 20)  |                  // pad_enable
                         (3u << 22)  |                  // pad_interval = 16 DW
                         (3u << 25)),                   // pad_amount  = 4 DW
                   (int)(((unsigned)C & 0xFFFFu) << 16),// tensor_dim0 lo
                   (int)(((unsigned)O & 0xFFFFu) << 16),// dim0 hi | tensor_dim1 lo
                   (int)(32u << 16),                    // dim1 hi | tile_dim0=32
                   (int)64,                             // tile_dim1=64, tile_dim2=0
                   (int)C,                              // tensor_dim0_stride lo
                   0, 0 };                              // stride hi, dim1_stride
      i32x4 g2 = { 0, 0, 0, 0 };
      i32x4 g3 = { 0, 0, 0, 0 };
#  if defined(__clang_major__) && __clang_major__ >= 23
      i32x8 g4 = { 0, 0, 0, 0, 0, 0, 0, 0 };
      __builtin_amdgcn_tensor_load_to_lds(g0, g1, g2, g3, g4, 0);
#  else
      __builtin_amdgcn_tensor_load_to_lds(g0, g1, g2, g3, 0);
#  endif
    }
#else
    for (int i = tid; i < 2048; i += 256) {
      const int c = i & 31, n = i >> 5;
      const int gn = n0 + n;
      sB[buf][n][c] = (gn < O) ? wt[(size_t)gn * C + c0 + c] : (bf16)0.f;
    }
#endif
  };

  v8f acc0 = {}, acc1 = {};
  const int m  = lane & 15;
  const int kb = (lane < 16) ? 0 : 8;

  stageA(0, 0);
  stageB(0, 0);
  int cur = 0;
  for (int ks = 0; ks < nk; ++ks) {
#if HAVE_TDM
    if (wv == 0) __builtin_amdgcn_s_wait_tensorcnt(0);
#endif
    __syncthreads();                       // buf[cur] fully staged
    if (ks + 1 < nk) {                     // overlap next-step staging
      stageA(ks + 1, cur ^ 1);
      stageB(ks + 1, cur ^ 1);
    }
    union { v16bf v; v8bf h[2]; } fb, fa0, fa1;
    fb.h[0]  = *(const v8bf*)&sB[cur][nwv * 16 + m][kb];
    fb.h[1]  = *(const v8bf*)&sB[cur][nwv * 16 + m][16 + kb];
    fa0.h[0] = *(const v8bf*)&sA[cur][mwv * 32 + m][kb];
    fa0.h[1] = *(const v8bf*)&sA[cur][mwv * 32 + m][16 + kb];
    fa1.h[0] = *(const v8bf*)&sA[cur][mwv * 32 + 16 + m][kb];
    fa1.h[1] = *(const v8bf*)&sA[cur][mwv * 32 + 16 + m][16 + kb];
    acc0 = __builtin_amdgcn_wmma_f32_16x16x32_bf16(false, fa0.v, false, fb.v,
                                                   (short)0, acc0, false, false);
    acc1 = __builtin_amdgcn_wmma_f32_16x16x32_bf16(false, fa1.v, false, fb.v,
                                                   (short)0, acc1, false, false);
    cur ^= 1;
  }

  // Fused epilogue: demod * acc + noise + bias, leaky ReLU.
  const float nw   = noise_w_all[noise_idx];
  const int   ncol = n0 + nwv * 16 + (lane & 15);
  if (ncol < O) {
    const float dd = d[b * O + ncol];
    const float bs = bias[ncol];
    float* outc = xout + ((size_t)(b * O + ncol)) * HW;
    const float* nz = noise + (size_t)b * HW;
#pragma unroll
    for (int r = 0; r < 8; ++r) {
      int p = p0 + mwv * 32 + r + ((lane < 16) ? 0 : 8);
      if (p < HW)      outc[p] = leaky02(acc0[r] * dd + nw * nz[p] + bs);
      p += 16;
      if (p < HW)      outc[p] = leaky02(acc1[r] * dd + nw * nz[p] + bs);
    }
  }
}

// ---------------------------------------------------------------------------
// toRGB (1x1 modulated conv, no demod) + skip upsample-add + optional tanh.
// ---------------------------------------------------------------------------
__global__ void torgb_kernel(const float* __restrict__ x,     // [4,C,H,W]
                             const float* __restrict__ rgbw,  // [3,C]
                             const float* __restrict__ srgb,  // [4,C]
                             const float* __restrict__ prev,  // [4,3,H/2,W/2] or null
                             float* __restrict__ out,         // [4,3,H,W]
                             int C, int H, int W, int has_prev, int do_tanh)
{
  const int idx = blockIdx.x * blockDim.x + threadIdx.x;
  const int HW = H * W;
  if (idx >= 4 * HW) return;
  const int b = idx / HW, p = idx % HW;
  const float scale = rsqrtf((float)C);
  const float* xb = x + (size_t)b * C * HW + p;
  const float* sb = srgb + b * C;
  float a0 = 0.f, a1 = 0.f, a2 = 0.f;
  for (int c = 0; c < C; ++c) {
    const float xv = xb[(size_t)c * HW];
    const float sv = sb[c] * scale * xv;
    a0 += rgbw[c] * sv;
    a1 += rgbw[C + c] * sv;
    a2 += rgbw[2 * C + c] * sv;
  }
  if (has_prev) {
    const int Hp = H >> 1, Wp = W >> 1;
    const int oy = p / W, ox = p % W;
    const float sy = oy * 0.5f - 0.25f, sx = ox * 0.5f - 0.25f;
    int y0 = (int)floorf(sy), x0 = (int)floorf(sx);
    const float wy = sy - (float)y0, wx = sx - (float)x0;
    int y1 = iclamp(y0 + 1, 0, Hp - 1), x1 = iclamp(x0 + 1, 0, Wp - 1);
    y0 = iclamp(y0, 0, Hp - 1); x0 = iclamp(x0, 0, Wp - 1);
    const float* pb = prev + (size_t)b * 3 * Hp * Wp;
    float add[3];
#pragma unroll
    for (int o = 0; o < 3; ++o) {
      const float* pl = pb + (size_t)o * Hp * Wp;
      const float v00 = pl[y0 * Wp + x0], v01 = pl[y0 * Wp + x1];
      const float v10 = pl[y1 * Wp + x0], v11 = pl[y1 * Wp + x1];
      add[o] = (1.f - wy) * ((1.f - wx) * v00 + wx * v01) +
               wy        * ((1.f - wx) * v10 + wx * v11);
    }
    a0 += add[0]; a1 += add[1]; a2 += add[2];
  }
  if (do_tanh) { a0 = tanhf(a0); a1 = tanhf(a1); a2 = tanhf(a2); }
  float* ob = out + (size_t)b * 3 * HW + p;
  ob[0] = a0; ob[HW] = a1; ob[2 * HW] = a2;
}

// ---------------------------------------------------------------------------
// Host orchestration.
// ---------------------------------------------------------------------------
extern "C" void kernel_launch(void* const* d_in, const int* in_sizes, int n_in,
                              void* d_out, int out_size, void* d_ws, size_t ws_size,
                              hipStream_t stream)
{
  (void)in_sizes; (void)n_in; (void)out_size; (void)ws_size;

  const float* z     = (const float*)d_in[0];
  const float* map_w = (const float*)d_in[1];
  const float* map_b = (const float*)d_in[2];
  const float* cnst  = (const float*)d_in[3];
  const float* conv_w[7]; const float* mod_w[7]; const float* mod_b[7];
  const float* bias_b[7]; const float* rgb_w[7]; const float* rgb_mw[7];
  const float* rgb_mb[7]; const float* noises[7];
  for (int i = 0; i < 7; ++i) {
    conv_w[i] = (const float*)d_in[4 + i];
    mod_w[i]  = (const float*)d_in[11 + i];
    mod_b[i]  = (const float*)d_in[18 + i];
    bias_b[i] = (const float*)d_in[26 + i];
    rgb_w[i]  = (const float*)d_in[33 + i];
    rgb_mw[i] = (const float*)d_in[40 + i];
    rgb_mb[i] = (const float*)d_in[47 + i];
    noises[i] = (const float*)d_in[54 + i];
  }
  const float* noise_w = (const float*)d_in[25];

  static const int INS[7]  = {512, 512, 512, 512, 512, 512, 256};
  static const int OUTS[7] = {512, 512, 512, 512, 512, 256, 128};

  // Workspace carve-up.
  char* wsp = (char*)d_ws;
  auto alloc = [&](size_t nbytes) -> char* {
    char* p = wsp;
    wsp += (nbytes + 255) & ~(size_t)255;
    return p;
  };
  float* wlat   = (float*)alloc(2048 * sizeof(float));  // mapped latents [4,512]
  float* sbuf   = (float*)alloc(2048 * sizeof(float));  // conv style s
  float* dbuf   = (float*)alloc(2048 * sizeof(float));  // demod d
  float* srgbb  = (float*)alloc(2048 * sizeof(float));  // rgb style
  float* wsq    = (float*)alloc((size_t)512 * 512 * sizeof(float));
  bf16*  mapwbf = (bf16*)alloc((size_t)8 * 512 * 512 * sizeof(bf16));
  bf16*  mwbf   = (bf16*)alloc((size_t)512 * 512 * sizeof(bf16));
  bf16*  wrepb  = (bf16*)alloc((size_t)9 * 512 * 512 * sizeof(bf16));
  const size_t XMME  = (size_t)4 * 256 * 256 * 256;     // max modulated input elems
  const size_t OUTME = (size_t)4 * 128 * 256 * 256;     // max conv output elems
  bf16*  xmbuf = (bf16*)alloc(XMME * sizeof(bf16));
  float* bufA  = (float*)alloc(OUTME * sizeof(float));
  float* bufB  = (float*)alloc(OUTME * sizeof(float));
  float* rgbPing = (float*)alloc((size_t)4 * 3 * 256 * 256 * sizeof(float));
  float* rgbPong = (float*)alloc((size_t)4 * 3 * 256 * 256 * sizeof(float));

  // 1) Mapping network: repack weights to bf16 once, then WMMA MLP.
  f32_to_bf16_kernel<<<(8 * 512 * 512 + 255) / 256, 256, 0, stream>>>(
      map_w, mapwbf, 8 * 512 * 512);
  mapping_wmma<<<1, 512, 0, stream>>>(z, mapwbf, map_b, wlat);

  float* xcur = nullptr;          // previous block conv output (f32)
  int Hc = 4, Wc = 4;
  const float* rgbprev = nullptr;

  for (int bi = 0; bi < 7; ++bi) {
    const int C = INS[bi], O = OUTS[bi];

    // Conv style (bf16 B), demod factors, bf16 weight repack.
    f32_to_bf16_kernel<<<(C * 512 + 255) / 256, 256, 0, stream>>>(
        mod_w[bi], mwbf, C * 512);
    style_wmma<<<(C + 63) / 64, 128, 0, stream>>>(wlat, mwbf, mod_b[bi], sbuf, C);
    wsq_kernel<<<(O * C + 255) / 256, 256, 0, stream>>>(conv_w[bi], wsq, O * C);
    demod_kernel<<<(4 * O + 255) / 256, 256, 0, stream>>>(wsq, sbuf, dbuf, C, O);
    wrepack_kernel<<<(9 * O * C + 255) / 256, 256, 0, stream>>>(conv_w[bi], wrepb, O, C);

    // Modulated bf16 input: broadcast const (block 0) or fused upsample.
    int H, W;
    if (bi == 0) {
      H = 4; W = 4;
      bcastmod_kernel<<<(4 * 512 * 16 + 255) / 256, 256, 0, stream>>>(
          cnst, sbuf, xmbuf, 512, 16);
    } else {
      H = 2 * Hc; W = 2 * Wc;
      const size_t total = (size_t)4 * C * H * W;
      upmod_kernel<<<(unsigned)((total + 255) / 256), 256, 0, stream>>>(
          xcur, sbuf, xmbuf, C, Hc, Wc);
    }

    // Implicit-GEMM conv (WMMA + TDM) with fused epilogue.
    float* xout = (xcur == bufA) ? bufB : bufA;
    dim3 grid((H * W + 63) / 64, (O + 63) / 64, 4);
    modconv3x3_wmma<<<grid, 256, 0, stream>>>(
        xmbuf, wrepb, dbuf, noises[bi], noise_w, bi, bias_b[bi],
        xout, C, O, H, W);

    // toRGB style (WMMA, bf16 B) + toRGB with fused skip-upsample-add (+tanh).
    f32_to_bf16_kernel<<<(O * 512 + 255) / 256, 256, 0, stream>>>(
        rgb_mw[bi], mwbf, O * 512);
    style_wmma<<<(O + 63) / 64, 128, 0, stream>>>(wlat, mwbf, rgb_mb[bi], srgbb, O);
    float* rgbout = (bi == 6) ? (float*)d_out : ((bi & 1) ? rgbPong : rgbPing);
    torgb_kernel<<<(4 * H * W + 255) / 256, 256, 0, stream>>>(
        xout, rgb_w[bi], srgbb, rgbprev, rgbout, O, H, W,
        /*has_prev=*/(bi > 0) ? 1 : 0, /*do_tanh=*/(bi == 6) ? 1 : 0);

    rgbprev = rgbout;
    xcur = xout; Hc = H; Wc = W;
  }
}
